// TransducerJoint_9182640079287
// MI455X (gfx1250) — matrix operand order repaired and checked by reading, not compiled, not run on hardware
//
#include <hip/hip_runtime.h>

typedef __attribute__((ext_vector_type(16))) _Float16 v16h;
typedef __attribute__((ext_vector_type(8)))  _Float16 v8h;
typedef __attribute__((ext_vector_type(4)))  _Float16 v4h;
typedef __attribute__((ext_vector_type(8)))  float    v8f;
typedef __attribute__((ext_vector_type(4)))  float    v4f;

#define B_     8
#define T_     256
#define U_     64
#define ENC_   512
#define PRED_  640
#define JOIN_  512
#define VOCAB_ 1024
#define M_     (B_ * T_ * U_)   // 131072

#define MT     128   // GEMM M tile
#define NT     128   // GEMM N tile
#define KT     32    // K per WMMA step
#define LDA    40    // padded LDS stride (halves); rows stay 16B aligned
#define LDB    40

// Branchless tanh on the hardware transcendental pipe:
// tanh(x) = 1 - 2/(exp2(2*log2(e)*x) + 1); saturates correctly at +/-inf.
__device__ __forceinline__ float fast_tanh(float x) {
    float e = __builtin_amdgcn_exp2f(x * 2.885390081777927f); // == e^(2x)
    return 1.0f - 2.0f * __builtin_amdgcn_rcpf(e + 1.0f);
}

// ---------------- Stage 1: projections  Y[m][n] = X[m][:].W[n][:] + bias[n]
// 1.4 GFLOP total (negligible); f32 keeps full precision feeding tanh.
__global__ void proj_kernel(const float* __restrict__ X,
                            const float* __restrict__ W,
                            const float* __restrict__ bias,
                            float* __restrict__ Y,
                            int M, int N, int K) {
    int gid = blockIdx.x * blockDim.x + threadIdx.x;
    if (gid >= M * N) return;
    int m = gid / N, n = gid - m * N;
    const v4f* x = (const v4f*)(X + (size_t)m * K);
    const v4f* w = (const v4f*)(W + (size_t)n * K);
    float acc = 0.f;
    int k4 = K >> 2;
    for (int k = 0; k < k4; ++k) {
        v4f a = x[k], b = w[k];
        acc += a.x * b.x + a.y * b.y + a.z * b.z + a.w * b.w;
    }
    Y[gid] = acc + bias[n];
}

// ---------------- Stage 2: A[bt][u][j] = f16(tanh(encP[bt][j] + predP[b,u][j]))
// Each of the 64M elements computed exactly once. encP (4MB) / predP (1MB) are
// L2-resident; the 128MB f16 write is ~5.5us at 23.3 TB/s.
__global__ __launch_bounds__(256) void tanh_stage_kernel(
        const float* __restrict__ encP,    // [B*T][JOIN]
        const float* __restrict__ predP,   // [B*U][JOIN]
        _Float16* __restrict__ A)          // [B*T*U][JOIN]
{
    int gid  = blockIdx.x * 256 + threadIdx.x;      // one per 4 elements
    size_t flat = (size_t)gid * 4;                  // < 64M
    int j  = (int)(flat & (JOIN_ - 1));
    int u  = (int)((flat >> 9) & (U_ - 1));
    int bt = (int)(flat >> 15);                     // / (JOIN_*U_)
    int b  = bt >> 8;                               // / T_
    v4f e = *(const v4f*)(encP  + (size_t)bt * JOIN_ + j);
    v4f p = *(const v4f*)(predP + ((size_t)(b * U_ + u)) * JOIN_ + j);
    v4h h;
    h.x = (_Float16)fast_tanh(e.x + p.x);
    h.y = (_Float16)fast_tanh(e.y + p.y);
    h.z = (_Float16)fast_tanh(e.z + p.z);
    h.w = (_Float16)fast_tanh(e.w + p.w);
    *(v4h*)(A + flat) = h;
}

// ---------------- Stage 3: out = A(f16) x W_out^T + b_out via v_wmma_f32_16x16x32_f16
// M=131072, N=1024, K=512. 128x128 block tile, 8 waves as 2(M) x 4(N);
// each wave owns 64x32 -> 8 accumulators, 8 WMMAs per k-step.
union HFrag { v16h v; v8h h[2]; };

__global__ __launch_bounds__(256)
void gemm_wmma_kernel(const _Float16* __restrict__ A,   // [M][JOIN]
                      const float* __restrict__ W,      // [VOCAB][JOIN]
                      const float* __restrict__ bias,   // [VOCAB]
                      float* __restrict__ out)          // [M][VOCAB]
{
    __shared__ __align__(32) _Float16 lA[MT * LDA];   // 128 x 32 f16 (+pad)
    __shared__ __align__(32) _Float16 lB[NT * LDB];   // 128 x 32 f16 (+pad)

    const int m_base = blockIdx.x * MT;
    const int n_base = blockIdx.y * NT;

    const int tid  = threadIdx.x;
    const int lane = tid & 31;            // wave32
    const int wave = tid >> 5;            // 0..7
    const int wm   = wave >> 2;           // 0..1  (64 rows each)
    const int wn   = wave & 3;            // 0..3  (32 cols each)

    v8f acc[4][2] = {};

    for (int k0 = 0; k0 < JOIN_; k0 += KT) {
        __syncthreads();
        // A tile: 128 rows x 32 k, already f16 in memory -> pure b128 copies.
        // 512 8-half units, 2 per thread.
        #pragma unroll
        for (int i = 0; i < 2; ++i) {
            int unit = i * 256 + tid;
            int row  = unit >> 2;
            int seg  = unit & 3;
            *(v8h*)&lA[row * LDA + seg * 8] =
                *(const v8h*)(A + (size_t)(m_base + row) * JOIN_ + k0 + seg * 8);
        }
        // B tile: 128 n x 32 k from f32 W_out, convert+pack to f16.
        // 1024 float4 units, 4 per thread.
        #pragma unroll
        for (int i = 0; i < 4; ++i) {
            int unit = i * 256 + tid;
            int n    = unit >> 3;
            int seg  = unit & 7;
            v4f wv = *(const v4f*)(W + (size_t)(n_base + n) * JOIN_ + k0 + seg * 4);
            v4h hv;
            hv.x = (_Float16)wv.x; hv.y = (_Float16)wv.y;
            hv.z = (_Float16)wv.z; hv.w = (_Float16)wv.w;
            *(v4h*)&lB[n * LDB + seg * 4] = hv;
        }
        if (k0 + KT < JOIN_) {
            // Pull next W_out k-slab toward L2/L0 (global_prefetch_b8).
            __builtin_prefetch(W + (size_t)(n_base + (tid >> 1)) * JOIN_ + k0 + KT
                                 + (tid & 1) * 16, 0, 1);
        }
        __syncthreads();

        // Fragment layouts (CDNA5 ISA §7.12.2, wave32):
        //  A 16x32 f16: lane<16 -> row=lane, K{0..7,16..23}; lane>=16 -> K{8..15,24..31}
        //  B 32x16 f16: lane&15 = column; lane<16 K{0..7,16..23}, lane>=16 K{8..15,24..31}
        const int hsel = lane >> 4;
        const int l15  = lane & 15;

        HFrag bf[2];
        #pragma unroll
        for (int ni = 0; ni < 2; ++ni) {
            int col = wn * 32 + ni * 16 + l15;
            bf[ni].h[0] = *(const v8h*)&lB[col * LDB + hsel * 8];
            bf[ni].h[1] = *(const v8h*)&lB[col * LDB + hsel * 8 + 16];
        }
        #pragma unroll
        for (int mi = 0; mi < 4; ++mi) {
            int row = wm * 64 + mi * 16 + l15;
            HFrag a;
            a.h[0] = *(const v8h*)&lA[row * LDA + hsel * 8];
            a.h[1] = *(const v8h*)&lA[row * LDA + hsel * 8 + 16];
            #pragma unroll
            for (int ni = 0; ni < 2; ++ni) {
                acc[mi][ni] = __builtin_amdgcn_wmma_f32_16x16x32_f16(
                    false, a.v, false, bf[ni].v, (short)0, acc[mi][ni],
                    false, false);
            }
        }
    }

    // Epilogue: C/D layout — VGPR r: lane<16 -> (M=r,N=lane); lane>=16 -> (M=r+8,N=lane-16)
    const int l15 = lane & 15;
    const int hi8 = (lane >> 4) * 8;
    #pragma unroll
    for (int ni = 0; ni < 2; ++ni) {
        int colg = n_base + wn * 32 + ni * 16 + l15;
        float bb = bias[colg];
        #pragma unroll
        for (int mi = 0; mi < 4; ++mi) {
            #pragma unroll
            for (int r = 0; r < 8; ++r) {
                size_t row = (size_t)m_base + wm * 64 + mi * 16 + r + hi8;
                out[row * VOCAB_ + colg] = acc[mi][ni][r] + bb;
            }
        }
    }
}

extern "C" void kernel_launch(void* const* d_in, const int* in_sizes, int n_in,
                              void* d_out, int out_size, void* d_ws, size_t ws_size,
                              hipStream_t stream) {
    const float* enc_out  = (const float*)d_in[0];  // [B,T,ENC]
    const float* pred_out = (const float*)d_in[1];  // [B,U,PRED]
    const float* W_enc    = (const float*)d_in[2];  // [JOIN,ENC]
    const float* b_enc    = (const float*)d_in[3];  // [JOIN]
    const float* W_pred   = (const float*)d_in[4];  // [JOIN,PRED]
    const float* b_pred   = (const float*)d_in[5];  // [JOIN]
    const float* W_out    = (const float*)d_in[6];  // [VOCAB,JOIN]
    const float* b_out    = (const float*)d_in[7];  // [VOCAB]
    float* out = (float*)d_out;

    // Workspace: encP 4MB f32 | predP 1MB f32 | A 128MB f16
    float*    encP  = (float*)d_ws;
    float*    predP = encP + (size_t)B_ * T_ * JOIN_;
    _Float16* Af16  = (_Float16*)(predP + (size_t)B_ * U_ * JOIN_);

    {
        int total = B_ * T_ * JOIN_;
        proj_kernel<<<(total + 255) / 256, 256, 0, stream>>>(
            enc_out, W_enc, b_enc, encP, B_ * T_, JOIN_, ENC_);
    }
    {
        int total = B_ * U_ * JOIN_;
        proj_kernel<<<(total + 255) / 256, 256, 0, stream>>>(
            pred_out, W_pred, b_pred, predP, B_ * U_, JOIN_, PRED_);
    }
    {
        int total4 = (M_ * JOIN_) / 4;   // 16M threads, 4 elems each
        tanh_stage_kernel<<<total4 / 256, 256, 0, stream>>>(encP, predP, Af16);
    }
    {
        dim3 grid(M_ / MT, VOCAB_ / NT);  // 1024 x 8
        gemm_wmma_kernel<<<grid, 256, 0, stream>>>(Af16, W_out, b_out, out);
    }
}
